// CoTrackerThreeOfflineModified_13305808683260
// MI455X (gfx1250) — matrix operand order repaired and checked by reading, not compiled, not run on hardware
//
#include <hip/hip_runtime.h>
#include <hip/hip_bf16.h>

// ---------------------------------------------------------------------------
// CoTracker correlation pipeline on gfx1250 (MI455X), bf16 WMMA everywhere.
// B=1, T=24, D=128, H=96, W=128, N=256, K=49 (7x7), 4 levels,
// MLP 2401->384 (GELU) ->256.  MLP1 uses async-to-LDS + TDM double buffering.
// ---------------------------------------------------------------------------

typedef __attribute__((ext_vector_type(16))) __bf16 bf16x16;
typedef __attribute__((ext_vector_type(8)))  __bf16 bf16x8;
typedef __attribute__((ext_vector_type(8)))  float  f32x8;
typedef __attribute__((ext_vector_type(4)))  unsigned int u32x4;
typedef __attribute__((ext_vector_type(8)))  int i32x8;
typedef __attribute__((ext_vector_type(4)))  int i32x4;

#define T_    24
#define D_    128
#define H0_   96
#define W0_   128
#define N_    256
#define KSUP  49
#define KPAD  64
#define VOLK  4096   /* 64*64 padded correlation-volume row */
#define HID   384
#define OUTC  256
#define BTN   (T_ * N_)   /* 6144 */

// ---------------------------------------------------------------------------
// Stage 1: L2-normalize fmaps (B,T,D,H,W) -> (t,h,w,d) channel-contiguous fp32
// ---------------------------------------------------------------------------
__global__ void k_normalize(const float* __restrict__ fmaps, float* __restrict__ out) {
    int th = blockIdx.x;               // t*H0 + h
    int w  = threadIdx.x;              // 0..127 == W0_
    const float* src = fmaps + (size_t)(th / H0_) * D_ * (H0_ * W0_)
                             + (size_t)(th % H0_) * W0_ + w;
    float ss = 0.f;
    #pragma unroll 4
    for (int d = 0; d < D_; ++d) { float v = src[(size_t)d * (H0_ * W0_)]; ss += v * v; }
    float inv = rsqrtf(fmaxf(ss, 1e-12f));
    float* dst = out + ((size_t)th * W0_ + w) * D_;
    #pragma unroll 4
    for (int d = 0; d < D_; ++d) dst[d] = src[(size_t)d * (H0_ * W0_)] * inv;
}

// 2x2 average pool, (t,h,w,d) layout
__global__ void k_pool(const float* __restrict__ in, float* __restrict__ out,
                       int Hi, int Wi, int Ho, int Wo) {
    size_t i = (size_t)blockIdx.x * blockDim.x + threadIdx.x;
    size_t total = (size_t)T_ * Ho * Wo * D_;
    if (i >= total) return;
    int d = (int)(i % D_); size_t r = i / D_;
    int w = (int)(r % Wo); r /= Wo;
    int h = (int)(r % Ho); int t = (int)(r / Ho);
    const float* b = in + (((size_t)t * Hi + 2 * h) * Wi + 2 * w) * D_ + d;
    out[i] = 0.25f * (b[0] + b[D_] + b[(size_t)Wi * D_] + b[(size_t)Wi * D_ + D_]);
}

// ---------------------------------------------------------------------------
// Weight prep: bf16, transposed, zero-padded to the 64*64 volume layout
// ---------------------------------------------------------------------------
__global__ void k_prep_w1t(const float* __restrict__ w1, __bf16* __restrict__ w1t) {
    int i = blockIdx.x * blockDim.x + threadIdx.x;      // n*4096 + (k*64+l)
    if (i >= HID * VOLK) return;
    int kk = i & (VOLK - 1); int n = i >> 12;
    int k = kk >> 6, l = kk & 63;
    float v = 0.f;
    if (k < KSUP && l < KSUP) v = w1[(k * KSUP + l) * HID + n];
    w1t[i] = (__bf16)v;
}

__global__ void k_prep_w2t(const float* __restrict__ w2, __bf16* __restrict__ w2t) {
    int i = blockIdx.x * blockDim.x + threadIdx.x;      // c*384 + j
    if (i >= OUTC * HID) return;
    int j = i % HID, c = i / HID;
    w2t[i] = (__bf16)w2[j * OUTC + c];
}

// ---------------------------------------------------------------------------
// Bilinear sampling (zero-padded), fp32 pyramid -> bf16 feature rows
// ---------------------------------------------------------------------------
__device__ __forceinline__ float bilin(const float* __restrict__ fm, int f,
                                       int Hl, int Wl, float px, float py, int d) {
    float x0f = floorf(px), y0f = floorf(py);
    int x0 = (int)x0f, y0 = (int)y0f;
    float tx = px - x0f, ty = py - y0f;
    float acc = 0.f;
    if (x0 >= 0 && x0 < Wl && y0 >= 0 && y0 < Hl)
        acc += (1.f - tx) * (1.f - ty) * fm[(((size_t)f * Hl + y0) * Wl + x0) * D_ + d];
    if (x0 + 1 >= 0 && x0 + 1 < Wl && y0 >= 0 && y0 < Hl)
        acc += tx * (1.f - ty) * fm[(((size_t)f * Hl + y0) * Wl + x0 + 1) * D_ + d];
    if (x0 >= 0 && x0 < Wl && y0 + 1 >= 0 && y0 + 1 < Hl)
        acc += (1.f - tx) * ty * fm[(((size_t)f * Hl + y0 + 1) * Wl + x0) * D_ + d];
    if (x0 + 1 >= 0 && x0 + 1 < Wl && y0 + 1 >= 0 && y0 + 1 < Hl)
        acc += tx * ty * fm[(((size_t)f * Hl + y0 + 1) * Wl + x0 + 1) * D_ + d];
    return acc;
}

// track-support features at the query frame: tf[n][k(64 pad)][d] bf16
__global__ void k_tf(const float* __restrict__ fm, const float* __restrict__ qcoords,
                     const int* __restrict__ qframes, __bf16* __restrict__ tf,
                     int Hl, int Wl, float sc) {
    int blk = blockIdx.x;                 // n*64 + k
    int n = blk >> 6, k = blk & 63, d = threadIdx.x;
    __bf16* dst = tf + (size_t)blk * D_;
    if (k >= KSUP) { dst[d] = (__bf16)0.f; return; }
    float px = qcoords[n * 2 + 0] * sc + (float)(k / 7 - 3);
    float py = qcoords[n * 2 + 1] * sc + (float)(k % 7 - 3);
    dst[d] = (__bf16)bilin(fm, qframes[n], Hl, Wl, px, py, d);
}

// correlation features per frame: cf[t*N+n][k(64 pad)][d] bf16
__global__ void k_cf(const float* __restrict__ fm, const float* __restrict__ coords,
                     __bf16* __restrict__ cf, int Hl, int Wl, float sc) {
    int blk = blockIdx.x;                 // (t*N+n)*64 + k
    int k = blk & 63, tn = blk >> 6, d = threadIdx.x;
    __bf16* dst = cf + (size_t)blk * D_;
    if (k >= KSUP) { dst[d] = (__bf16)0.f; return; }
    int t = tn >> 8;                      // N_ == 256
    float px = coords[tn * 2 + 0] * sc + (float)(k / 7 - 3);
    float py = coords[tn * 2 + 1] * sc + (float)(k % 7 - 3);
    dst[d] = (__bf16)bilin(fm, t, Hl, Wl, px, py, d);
}

// ---------------------------------------------------------------------------
// WMMA fragment loaders (ISA 7.12.2 layouts).
// A 16x32 bf16: lane m=L&15, hi=L>>4; v0..3 <- K[k0+hi*8 .. +7], v4..7 <- +16.
// B 32x16 bf16 (from transposed operand rows): lane n=L&15, hi=L>>4;
//   v0..3 <- K[k0+hi*8 .. +7], v4..7 <- +16 (same as A).
// ---------------------------------------------------------------------------
__device__ __forceinline__ bf16x16 load_fragA(const __bf16* __restrict__ base,
                                              int ld, int row0, int k0, int lane) {
    int m = lane & 15, hi = (lane >> 4) & 1;
    const __bf16* p = base + (size_t)(row0 + m) * ld + (k0 + hi * 8);
    bf16x8 lo = *reinterpret_cast<const bf16x8*>(p);
    bf16x8 hh = *reinterpret_cast<const bf16x8*>(p + 16);
    return __builtin_shufflevector(lo, hh, 0,1,2,3,4,5,6,7,8,9,10,11,12,13,14,15);
}

__device__ __forceinline__ bf16x16 load_fragB(const __bf16* __restrict__ baseT,
                                              int ld, int n0, int k0, int lane) {
    int n = lane & 15, hi = (lane >> 4) & 1;
    const __bf16* p = baseT + (size_t)(n0 + n) * ld + (k0 + hi * 8);
    bf16x8 lo = *reinterpret_cast<const bf16x8*>(p);
    bf16x8 hh = *reinterpret_cast<const bf16x8*>(p + 16);
    return __builtin_shufflevector(lo, hh, 0,1,2,3,4,5,6,7,8,9,10,11,12,13,14,15);
}

// LDS byte offset of a __shared__ object (flat addr low 32 bits == LDS offset)
__device__ __forceinline__ unsigned lds_off(const void* p) {
    return (unsigned)(uintptr_t)p;
}

// ---------------------------------------------------------------------------
// Correlation volume: per (t,n), vol[64x64] = cf(64x128) x tf(64x128)^T.
// 4 waves/block, one 16-row tile per wave, 4 n-tiles, 4 k-steps -> 16 WMMA.
// ---------------------------------------------------------------------------
__global__ void k_corr(const __bf16* __restrict__ cf, const __bf16* __restrict__ tf,
                       __bf16* __restrict__ vol) {
    int pid = blockIdx.x;                 // t*N + n
    int n = pid & (N_ - 1);
    int wave = threadIdx.x >> 5, lane = threadIdx.x & 31;
    const __bf16* A = cf + (size_t)pid * KPAD * D_;
    const __bf16* Bm = tf + (size_t)n * KPAD * D_;
    f32x8 zero = {0.f,0.f,0.f,0.f,0.f,0.f,0.f,0.f};
    f32x8 acc[4] = {zero, zero, zero, zero};
    #pragma unroll
    for (int ks = 0; ks < 4; ++ks) {
        bf16x16 a = load_fragA(A, D_, wave * 16, ks * 32, lane);
        #pragma unroll
        for (int j = 0; j < 4; ++j) {
            bf16x16 b = load_fragB(Bm, D_, j * 16, ks * 32, lane);
            acc[j] = __builtin_amdgcn_wmma_f32_16x16x32_bf16(
                false, a, false, b, (short)0, acc[j], false, false);
        }
    }
    __bf16* out = vol + (size_t)pid * VOLK;
    int hi = (lane >> 4) & 1, nl = lane & 15;
    #pragma unroll
    for (int j = 0; j < 4; ++j) {
        #pragma unroll
        for (int r = 0; r < 8; ++r) {
            out[(wave * 16 + r + hi * 8) * 64 + j * 16 + nl] = (__bf16)acc[j][r];
        }
    }
}

// ---------------------------------------------------------------------------
// MLP1: hidden(6144x384) = GELU(vol(6144x4096) @ w1t^T + b1), bf16 out.
// Block = 64(M) x 64(N), 4 waves.  Double-buffered LDS staging:
//   A tile (64x32 bf16) via per-lane global_load_async_to_lds_b128 (ASYNCcnt),
//   B tile (64x32 bf16 of w1t) via TDM tensor_load_to_lds (TENSORcnt, wave 0).
// ---------------------------------------------------------------------------
__global__ void k_mlp1(const __bf16* __restrict__ vol, const __bf16* __restrict__ w1t,
                       const float* __restrict__ b1, __bf16* __restrict__ hidden) {
    __shared__ __align__(16) __bf16 sA[2][64 * 32];
    __shared__ __align__(16) __bf16 sB[2][64 * 32];

    int tid  = threadIdx.x;
    int wave = tid >> 5, lane = tid & 31;
    int rowBlk = blockIdx.x * 64;          // 64 M rows of this block
    int colBlk = blockIdx.y * 64;          // 64 N cols of this block
    unsigned ldsA[2] = { lds_off(&sA[0][0]), lds_off(&sA[1][0]) };
    unsigned ldsB[2] = { lds_off(&sB[0][0]), lds_off(&sB[1][0]) };

    // A tile: 4 KB = 256 x 16B chunks; 128 threads x 2 async b128 copies.
    auto stageA = [&](int ks, int buf) {
        #pragma unroll
        for (int i = 0; i < 2; ++i) {
            int chunk = tid * 2 + i;                   // 0..255
            int row = chunk >> 2, c16 = chunk & 3;     // row 0..63, 16B chunk
            unsigned long long g = (unsigned long long)(uintptr_t)
                (vol + (size_t)(rowBlk + row) * VOLK + ks * 32) + (unsigned)(c16 * 16);
            unsigned l = ldsA[buf] + (unsigned)(row * 64 + c16 * 16);
            asm volatile("global_load_async_to_lds_b128 %0, %1, off"
                         :: "v"(l), "v"(g) : "memory");
        }
    };
    // B tile via Tensor Data Mover: 2D tile 32(elems) x 64(rows), 2B elements,
    // row stride 4096 elements, from w1t[colBlk..colBlk+63][ks*32..+31].
    auto stageB = [&](int ks, int buf) {
        unsigned long long g = (unsigned long long)(uintptr_t)
            (w1t + (size_t)colBlk * VOLK + ks * 32);
        u32x4 g0;
        g0[0] = 1u;                                            // count=1
        g0[1] = ldsB[buf];                                     // lds_addr
        g0[2] = (unsigned)g;                                   // global_addr[31:0]
        g0[3] = (unsigned)((g >> 32) & 0x1FFFFFFull) | 0x80000000u; // [56:32] | type=2
        i32x8 g1;
        g1[0] = 0x00010000;                 // workgroup_mask=0, data_size=1 (2B)
        g1[1] = (int)(0x1000u << 16);       // tensor_dim0 = 4096 (bits 79:48 lo16)
        g1[2] = (int)(((unsigned)HID) << 16); // tensor_dim0 hi=0 | tensor_dim1=384
        g1[3] = 32 << 16;                   // tensor_dim1 hi=0 | tile_dim0=32
        g1[4] = 64;                         // tile_dim1=64, tile_dim2=0
        g1[5] = 4096;                       // tensor_dim0_stride[31:0]
        g1[6] = 0;                          // stride hi, tensor_dim1_stride lo
        g1[7] = 0;
        i32x4 z4 = {0, 0, 0, 0};
        i32x8 z8 = {0, 0, 0, 0, 0, 0, 0, 0};
        __builtin_amdgcn_tensor_load_to_lds(g0, g1, z4, z4, z8, 0);
    };

    f32x8 zero = {0.f,0.f,0.f,0.f,0.f,0.f,0.f,0.f};
    f32x8 acc[4] = {zero, zero, zero, zero};

    const int KSTEPS = VOLK / 32;          // 128
    stageA(0, 0);
    if (wave == 0) stageB(0, 0);

    #pragma unroll 2
    for (int ks = 0; ks < KSTEPS; ++ks) {
        int cur = ks & 1, nxt = cur ^ 1;
        if (ks + 1 < KSTEPS) {
            stageA(ks + 1, nxt);
            if (wave == 0) stageB(ks + 1, nxt);
            asm volatile("s_wait_asynccnt 0x2" ::: "memory");
            if (wave == 0) __builtin_amdgcn_s_wait_tensorcnt((short)1);
        } else {
            asm volatile("s_wait_asynccnt 0x0" ::: "memory");
            if (wave == 0) __builtin_amdgcn_s_wait_tensorcnt((short)0);
        }
        __syncthreads();                   // staged tiles visible to all waves

        // consume: A fragment from LDS (per-wave 16 rows), 4 B fragments
        {
            int m = lane & 15, hi = (lane >> 4) & 1;
            const __bf16* pa = &sA[cur][(wave * 16 + m) * 32 + hi * 8];
            bf16x8 alo = *reinterpret_cast<const bf16x8*>(pa);
            bf16x8 ahi = *reinterpret_cast<const bf16x8*>(pa + 16);
            bf16x16 a = __builtin_shufflevector(alo, ahi,
                0,1,2,3,4,5,6,7,8,9,10,11,12,13,14,15);
            #pragma unroll
            for (int j = 0; j < 4; ++j) {
                const __bf16* pb = &sB[cur][(j * 16 + m) * 32 + hi * 8];
                bf16x8 blo = *reinterpret_cast<const bf16x8*>(pb);
                bf16x8 bhi = *reinterpret_cast<const bf16x8*>(pb + 16);
                bf16x16 b = __builtin_shufflevector(blo, bhi,
                    0,1,2,3,4,5,6,7,8,9,10,11,12,13,14,15);
                acc[j] = __builtin_amdgcn_wmma_f32_16x16x32_bf16(
                    false, a, false, b, (short)0, acc[j], false, false);
            }
        }
        __syncthreads();                   // protect buffers before re-staging
    }

    int hi = (lane >> 4) & 1, nl = lane & 15;
    #pragma unroll
    for (int j = 0; j < 4; ++j) {
        #pragma unroll
        for (int r = 0; r < 8; ++r) {
            int m = rowBlk + wave * 16 + r + hi * 8;
            int c = colBlk + j * 16 + nl;
            float x = acc[j][r] + b1[c];
            float g = 0.5f * x * (1.0f + erff(x * 0.70710678118654752f));
            hidden[(size_t)m * HID + c] = (__bf16)g;
        }
    }
}

// ---------------------------------------------------------------------------
// MLP2: out-slice(6144x256) = hidden(6144x384) @ w2t^T + b2, fp32 into d_out.
// ---------------------------------------------------------------------------
__global__ void k_mlp2(const __bf16* __restrict__ hidden, const __bf16* __restrict__ w2t,
                       const float* __restrict__ b2, float* __restrict__ out, int lvl) {
    int wave = threadIdx.x >> 5, lane = threadIdx.x & 31;
    int row0 = blockIdx.x * 64 + wave * 16;
    int col0 = blockIdx.y * 64;
    f32x8 zero = {0.f,0.f,0.f,0.f,0.f,0.f,0.f,0.f};
    f32x8 acc[4] = {zero, zero, zero, zero};
    #pragma unroll
    for (int ks = 0; ks < HID / 32; ++ks) {            // 12 k-steps
        bf16x16 a = load_fragA(hidden, HID, row0, ks * 32, lane);
        #pragma unroll
        for (int j = 0; j < 4; ++j) {
            bf16x16 b = load_fragB(w2t, HID, col0 + j * 16, ks * 32, lane);
            acc[j] = __builtin_amdgcn_wmma_f32_16x16x32_bf16(
                false, a, false, b, (short)0, acc[j], false, false);
        }
    }
    int hi = (lane >> 4) & 1, nl = lane & 15;
    #pragma unroll
    for (int j = 0; j < 4; ++j) {
        #pragma unroll
        for (int r = 0; r < 8; ++r) {
            int m = row0 + r + hi * 8;
            int c = col0 + j * 16 + nl;
            out[(size_t)m * (4 * OUTC) + lvl * OUTC + c] = acc[j][r] + b2[c];
        }
    }
}

// ---------------------------------------------------------------------------
// Host-side orchestration (graph-capture safe: only kernel launches on stream)
// ---------------------------------------------------------------------------
extern "C" void kernel_launch(void* const* d_in, const int* in_sizes, int n_in,
                              void* d_out, int out_size, void* d_ws, size_t ws_size,
                              hipStream_t stream) {
    const float* fmaps   = (const float*)d_in[0];
    const float* coords  = (const float*)d_in[1];
    const float* qcoords = (const float*)d_in[2];
    const int*   qframes = (const int*)d_in[3];
    const float* w1      = (const float*)d_in[4];
    const float* b1      = (const float*)d_in[5];
    const float* w2      = (const float*)d_in[6];
    const float* b2      = (const float*)d_in[7];
    float* out = (float*)d_out;

    static const int Hl[4] = {96, 48, 24, 12};
    static const int Wl[4] = {128, 64, 32, 16};

    char* ws = (char*)d_ws;
    size_t o = 0;
    float* pyr[4];
    for (int l = 0; l < 4; ++l) {
        pyr[l] = (float*)(ws + o);
        o += (size_t)T_ * Hl[l] * Wl[l] * D_ * sizeof(float);
    }
    __bf16* tf     = (__bf16*)(ws + o); o += (size_t)N_ * KPAD * D_ * 2;
    __bf16* cf     = (__bf16*)(ws + o); o += (size_t)BTN * KPAD * D_ * 2;
    __bf16* vol    = (__bf16*)(ws + o); o += (size_t)BTN * VOLK * 2;
    __bf16* hidden = (__bf16*)(ws + o); o += (size_t)BTN * HID * 2;
    __bf16* w1t    = (__bf16*)(ws + o); o += (size_t)HID * VOLK * 2;
    __bf16* w2t    = (__bf16*)(ws + o); o += (size_t)OUTC * HID * 2;
    (void)o; (void)in_sizes; (void)n_in; (void)out_size; (void)ws_size;

    // Pyramid
    k_normalize<<<T_ * H0_, 128, 0, stream>>>(fmaps, pyr[0]);
    for (int l = 0; l < 3; ++l) {
        size_t tot = (size_t)T_ * Hl[l + 1] * Wl[l + 1] * D_;
        k_pool<<<(unsigned)((tot + 255) / 256), 256, 0, stream>>>(
            pyr[l], pyr[l + 1], Hl[l], Wl[l], Hl[l + 1], Wl[l + 1]);
    }
    // Weight prep (bf16, transposed, zero-padded)
    k_prep_w1t<<<(HID * VOLK + 255) / 256, 256, 0, stream>>>(w1, w1t);
    k_prep_w2t<<<(OUTC * HID + 255) / 256, 256, 0, stream>>>(w2, w2t);

    for (int l = 0; l < 4; ++l) {
        float sc = 1.0f / (float)(1 << l);
        k_tf<<<N_ * KPAD, 128, 0, stream>>>(pyr[l], qcoords, qframes, tf, Hl[l], Wl[l], sc);
        k_cf<<<BTN * KPAD, 128, 0, stream>>>(pyr[l], coords, cf, Hl[l], Wl[l], sc);
        k_corr<<<BTN, 128, 0, stream>>>(cf, tf, vol);
        k_mlp1<<<dim3(BTN / 64, HID / 64), 128, 0, stream>>>(vol, w1t, b1, hidden);
        k_mlp2<<<dim3(BTN / 64, OUTC / 64), 128, 0, stream>>>(hidden, w2t, b2, out, l);
    }
}
